// BSpline_76158360093294
// MI455X (gfx1250) — compile-verified, matrix operand order checked by reading, not run on hardware
//
#include <hip/hip_runtime.h>

typedef __attribute__((ext_vector_type(16))) _Float16 v16h;
typedef __attribute__((ext_vector_type(8)))  float    v8f;
typedef __attribute__((ext_vector_type(4)))  unsigned u32x4;

namespace {
constexpr int ROWS  = 65536;
constexpr int CH    = 64;            // input channels
constexpr int COEF  = 12;            // basis funcs per channel
constexpr int KDIM  = CH * COEF;     // 768 (GEMM K)
constexpr int NDIM  = 64;            // output channels (GEMM N)
constexpr int WAVES = 8;
constexpr int MW    = 16;            // rows per wave (WMMA M)
constexpr int MB    = WAVES * MW;    // 128 rows per block
constexpr int QCH   = 16;            // channels per quarter
constexpr int QK    = QCH * COEF;    // 192 K per quarter
constexpr int NQ    = CH / QCH;      // 4 quarters
constexpr int KT_PER_Q = QK / 32;    // 6 WMMA K-steps per quarter

constexpr int BT_HALVES = NDIM * KDIM;             // 49152 halves (96 KB)
constexpr int AW_HALVES = MW * QK;                 // 3072 halves per wave (6 KB)
constexpr size_t SMEM_BYTES =
    (size_t)(BT_HALVES + WAVES * AW_HALVES) * sizeof(_Float16);  // 147456 B
}

// knots[i] = clamp((i-3)/9, 0, 1) for N_KNOTS=16, degree=3 (computed, never indexed)
__device__ __forceinline__ float knotv(int i) {
  return fminf(fmaxf((float)(i - 3) * (1.0f / 9.0f), 0.0f), 1.0f);
}

// Windowed Cox-de Boor: only 4 of the 12 cubic basis functions are non-zero.
// Returns nv[0..3] = basis[s .. s+3]; s = active span index.
__device__ __forceinline__ int bspline4(float x, float nv[4]) {
  int s = (int)(x * 9.0f);
  if (s > 8) s = 8;                 // x == 1 falls into the last real span
  const int jj = s + 3;
  float left[3], right[3];
  nv[0] = 1.0f;
  #pragma unroll
  for (int d = 1; d <= 3; ++d) {
    left[d - 1]  = x - knotv(jj + 1 - d);
    right[d - 1] = knotv(jj + d) - x;
    float saved = 0.0f;
    #pragma unroll
    for (int r = 0; r < d; ++r) {
      const float den = right[r] + left[d - 1 - r];   // > 0 for interior spans
      const float tmp = nv[r] * __builtin_amdgcn_rcpf(den);
      nv[r] = saved + right[r] * tmp;
      saved = left[d - 1 - r] * tmp;
    }
    nv[d] = saved;
  }
  return s;
}

// One 16x32 f16 fragment (A: lane=M row; B: lane=N col). Half-lane group hl
// needs halves [hl*8, hl*8+8) and [hl*8+16, hl*8+24) -> two ds_load_b128.
union Frag { u32x4 q[2]; v16h h; };

__device__ __forceinline__ v16h load_frag(const _Float16* p, int hl) {
  Frag f;
  const u32x4* p4 = (const u32x4*)(p + hl * 8);   // 16B aligned by construction
  f.q[0] = p4[0];       // halves base+0 .. base+7   (VGPR 0-3 of fragment)
  f.q[1] = p4[2];       // halves base+16 .. base+23 (VGPR 4-7 of fragment)
  return f.h;
}

__global__ __launch_bounds__(256)
void bspline_kan_wmma(const float* __restrict__ X,
                      const float* __restrict__ CP,
                      float* __restrict__ OUT) {
  extern __shared__ _Float16 smem[];
  _Float16* Bt = smem;  // weights, N-major: Bt[n*KDIM + k] -> (k,k+1) contiguous
  const int tid  = threadIdx.x;
  const int wave = tid >> 5;
  const int lane = tid & 31;
  _Float16* Aw = smem + BT_HALVES + wave * AW_HALVES;  // per-wave strip [MW][QK]

  // Stage control points -> LDS f16 (CP flat index e = k*64 + n).
  for (int e = tid; e < KDIM * NDIM; e += 256) {
    const int k = e >> 6;
    const int n = e & 63;
    Bt[n * KDIM + k] = (_Float16)CP[e];
  }

  const int rowBase = blockIdx.x * MB + wave * MW;
  __builtin_prefetch(&X[(size_t)rowBase * CH], 0, 0);  // global_prefetch_b8

  v8f acc[4];
  const v8f vzero = {};
  #pragma unroll
  for (int nt = 0; nt < 4; ++nt) acc[nt] = vzero;

  const int idx = lane & 15;   // M index (A) / N index (B,C)
  const int hl  = lane >> 4;   // half-lane group

  __syncthreads();             // only barrier needed: Bt is written by all waves

  for (int q = 0; q < NQ; ++q) {
    // Clear this wave's strip with wide stores (wave-private; LDS is in-order
    // within a wave, so no barrier is needed between clear/eval/consume).
    {
      u32x4* az = (u32x4*)Aw;                       // 6144 B = 384 u32x4
      const u32x4 z4 = {0u, 0u, 0u, 0u};
      #pragma unroll
      for (int i = 0; i < 12; ++i) az[i * 32 + lane] = z4;
    }

    // Evaluate basis: 16 rows x 16 channels, 4 non-zero values per element.
    #pragma unroll 1
    for (int it = 0; it < (MW * QCH) / 32; ++it) {   // 8 iterations
      const int p  = it * 32 + lane;
      const int m  = p >> 4;                         // row within strip
      const int cl = p & 15;                         // channel within quarter
      float x = X[(size_t)(rowBase + m) * CH + (q * QCH + cl)];
      x = fminf(fmaxf(x, 0.0f), 1.0f);
      float nv[4];
      const int s = bspline4(x, nv);
      _Float16* dst = Aw + m * QK + cl * COEF + s;
      #pragma unroll
      for (int j = 0; j < 4; ++j) dst[j] = (_Float16)nv[j];
    }

    // 6 K-steps; batch A + 4 B fragments, then 4 back-to-back WMMAs.
    for (int kt = 0; kt < KT_PER_Q; ++kt) {
      const v16h fA = load_frag(Aw + idx * QK + kt * 32, hl);
      const int kg = q * QK + kt * 32;
      v16h fB[4];
      #pragma unroll
      for (int nt = 0; nt < 4; ++nt)
        fB[nt] = load_frag(Bt + (nt * 16 + idx) * KDIM + kg, hl);
      #pragma unroll
      for (int nt = 0; nt < 4; ++nt)
        acc[nt] = __builtin_amdgcn_wmma_f32_16x16x32_f16(
            false, fA, false, fB[nt], (short)0, acc[nt], false, false);
    }
  }

  // C/D layout: VGPR r holds (M = r + 8*hl, N = idx) per 16x16 f32 tile.
  #pragma unroll
  for (int nt = 0; nt < 4; ++nt) {
    const int n = nt * 16 + idx;
    #pragma unroll
    for (int r = 0; r < 8; ++r) {
      const int m = r + hl * 8;
      OUT[(size_t)(rowBase + m) * NDIM + n] = acc[nt][r];
    }
  }
}

extern "C" void kernel_launch(void* const* d_in, const int* in_sizes, int n_in,
                              void* d_out, int out_size, void* d_ws, size_t ws_size,
                              hipStream_t stream) {
  (void)in_sizes; (void)n_in; (void)out_size; (void)d_ws; (void)ws_size;
  const float* X  = (const float*)d_in[0];   // (65536, 64) f32
  const float* CP = (const float*)d_in[1];   // (64, 12, 64) f32
  float* OUT = (float*)d_out;                // (65536, 64) f32

  (void)hipFuncSetAttribute(reinterpret_cast<const void*>(&bspline_kan_wmma),
                            hipFuncAttributeMaxDynamicSharedMemorySize,
                            (int)SMEM_BYTES);

  dim3 grid(ROWS / MB);   // 512 blocks
  dim3 block(256);        // 8 wave32s
  bspline_kan_wmma<<<grid, block, SMEM_BYTES, stream>>>(X, CP, OUT);
}
